// LinearAttention_26225070309967
// MI455X (gfx1250) — matrix operand order repaired and checked by reading, not compiled, not run on hardware
//
#include <hip/hip_runtime.h>

typedef __attribute__((ext_vector_type(16))) __bf16 v16bf;
typedef __attribute__((ext_vector_type(8)))  float  v8f;
typedef __attribute__((ext_vector_type(4)))  unsigned int v4u;
typedef __attribute__((ext_vector_type(8)))  int    v8i;
typedef __attribute__((ext_vector_type(4)))  int    v4i;

#define NB     4
#define S_LEN  8192
#define DIN    1024
#define H      1024

#define ROW_STRIDE   80          // 64B tile row + 16B TDM pad -> conflict-free LDS
#define TILE1_BYTES  (32 * ROW_STRIDE)   // 32 cols x 32 k  (qkv weights)
#define TILE2_BYTES  (64 * ROW_STRIDE)   // 64 cols x 32 k  (wo)

union FragU { uint4 u[2]; v16bf v; };
union Pack8 { uint4 u; __bf16 h[8]; };

// ---------------- WMMA helper -----------------------------------------------

__device__ inline v8f wmma_bf16(v16bf a, v16bf b, v8f c) {
    return __builtin_amdgcn_wmma_f32_16x16x32_bf16(
        false, a, false, b, (short)0, c, false, false);
}

// A fragment: row-major [row][k], ld=1024 (caller passes klo = k0 + (lane>>4)*8)
__device__ inline v16bf load_a_frag(const __bf16* base, int row, int klo) {
    FragU t;
    const uint4* p = reinterpret_cast<const uint4*>(base + (size_t)row * DIN + klo);
    t.u[0] = p[0];   // K = klo .. klo+7
    t.u[1] = p[2];   // K = klo+16 .. klo+23
    return t.v;
}

// B fragment from an LDS-staged tile: row nl (= local output column), 80B stride;
// lane half lh selects K 8lh..8lh+7 and 16+8lh..16+8lh+7 within the 32-wide k-slice.
__device__ inline v16bf lds_b_frag(const char* tile, int nl, int lh) {
    FragU t;
    const uint4* p = reinterpret_cast<const uint4*>(tile + nl * ROW_STRIDE + lh * 16);
    t.u[0] = p[0];
    t.u[1] = p[2];
    return t.v;
}

// ---------------- TDM: async 2D bf16 tile -> LDS -----------------------------
// Tensor = wT[1024][1024] bf16; tile = tile_rows x 32 elements starting at
// element offset elem_off. Pad 4 DWORDs after every 16 DWORDs -> 80B LDS rows.
__device__ inline void tdm_load_tile(unsigned lds_off, const __bf16* gbase,
                                     long long elem_off, int tile_rows) {
    unsigned long long ga = (unsigned long long)(size_t)gbase +
                            (unsigned long long)elem_off * 2ull;
    v4u g0;
    g0.x = 1u;                                            // count=1, user mode
    g0.y = lds_off;                                       // LDS byte address
    g0.z = (unsigned)(ga & 0xffffffffu);                  // global_addr lo
    g0.w = (unsigned)((ga >> 32) & 0x01ffffffu) | (2u << 30);   // addr hi | type=2
    v8i g1;
    g1[0] = (int)((1u << 16) | (1u << 20) | (3u << 22) | (3u << 25));
             // data_size=2B | pad_enable | pad_interval=16dw | pad_amount=4dw
    g1[1] = (int)(DIN << 16);            // tensor_dim0 = 1024 (low16)
    g1[2] = (int)((H & 0xffff) << 16);   // tensor_dim0 hi=0 | tensor_dim1 lo=1024
    g1[3] = (int)(32u << 16);            // tensor_dim1 hi=0 | tile_dim0 = 32
    g1[4] = tile_rows;                   // tile_dim1 | tile_dim2=0
    g1[5] = DIN;                         // tensor_dim0_stride = 1024
    g1[6] = 0;
    g1[7] = 0;
    v4i g2 = {0, 0, 0, 0}, g3 = {0, 0, 0, 0};
#if defined(__clang_major__) && __clang_major__ >= 23
    v8i g4 = {0, 0, 0, 0, 0, 0, 0, 0};
    __builtin_amdgcn_tensor_load_to_lds(g0, g1, g2, g3, g4, 0);
#else
    __builtin_amdgcn_tensor_load_to_lds(g0, g1, g2, g3, 0);
#endif
}

// ---------------- prep kernels ----------------------------------------------

__global__ __launch_bounds__(256) void zero_f32(float* p, int n) {
    int i = blockIdx.x * 256 + threadIdx.x;
    if (i < n) p[i] = 0.0f;
}

__global__ __launch_bounds__(256) void cvt_x_bf16(const float* __restrict__ x,
                                                  __bf16* __restrict__ xb, size_t n8) {
    size_t i = (size_t)blockIdx.x * 256 + threadIdx.x;
    if (i >= n8) return;
    size_t base = i * 8;
    float4 a = *reinterpret_cast<const float4*>(x + base);
    float4 b = *reinterpret_cast<const float4*>(x + base + 4);
    Pack8 o;
    o.h[0] = (__bf16)a.x; o.h[1] = (__bf16)a.y; o.h[2] = (__bf16)a.z; o.h[3] = (__bf16)a.w;
    o.h[4] = (__bf16)b.x; o.h[5] = (__bf16)b.y; o.h[6] = (__bf16)b.z; o.h[7] = (__bf16)b.w;
    *reinterpret_cast<uint4*>(xb + base) = o.u;
}

// 32x32 LDS tile transpose + fp32->bf16: wt[n][k] = w[k][n]
__global__ __launch_bounds__(256) void transpose_cvt(const float* __restrict__ w,
                                                     __bf16* __restrict__ wt) {
    __shared__ float tile[32][33];
    int tx = threadIdx.x, ty = threadIdx.y;            // blockDim = (32, 8)
    int bx = blockIdx.x * 32, by = blockIdx.y * 32;
#pragma unroll
    for (int j = 0; j < 32; j += 8)
        tile[ty + j][tx] = w[(size_t)(by + ty + j) * H + bx + tx];
    __syncthreads();
#pragma unroll
    for (int j = 0; j < 32; j += 8)
        wt[(size_t)(bx + ty + j) * DIN + by + tx] = (__bf16)tile[tx][ty + j];
}

// ---------------- kernel 1: fused QKV + elu feature map + seq reductions ----
// Block = 8 waves = 128 rows x 32 cols macro-tile; weight tiles arrive in LDS
// via double-buffered TDM driven by wave 0.

__global__ __launch_bounds__(256) void qkv_wmma(
    const __bf16* __restrict__ xb,
    const __bf16* __restrict__ wqT, const __bf16* __restrict__ wkT,
    const __bf16* __restrict__ wvT,
    const float* __restrict__ bq, const float* __restrict__ bk,
    const float* __restrict__ bv,
    __bf16* __restrict__ qfeat, float* __restrict__ kvred, float* __restrict__ kred)
{
    __shared__ alignas(16) char smem[2 * 3 * TILE1_BYTES];
    const int lane = threadIdx.x & 31;
    const int wid  = threadIdx.x >> 5;
    const int rg = blockIdx.x >> 5;          // 256 row groups of 128
    const int cg = blockIdx.x & 31;          // 32 col groups of 32
    const int m0 = (rg << 7) + (wid << 4);
    const int lm = lane & 15, lh = lane >> 4;
    const int arow = m0 + lm;
    const int col0 = cg << 5;
    const unsigned lds_base = (unsigned)(size_t)&smem[0];   // flat->LDS: addr[31:0]

    v8f cq[2], ck[2], cv[2];
#pragma unroll
    for (int t = 0; t < 2; ++t) {
        cq[t] = (v8f){0,0,0,0,0,0,0,0};
        ck[t] = (v8f){0,0,0,0,0,0,0,0};
        cv[t] = (v8f){0,0,0,0,0,0,0,0};
    }

    if (wid == 0) {
        long long off0 = (long long)col0 * DIN;
        tdm_load_tile(lds_base + 0 * TILE1_BYTES, wqT, off0, 32);
        tdm_load_tile(lds_base + 1 * TILE1_BYTES, wkT, off0, 32);
        tdm_load_tile(lds_base + 2 * TILE1_BYTES, wvT, off0, 32);
    }

    for (int k0 = 0; k0 < DIN; k0 += 32) {
        const int cur = (k0 >> 5) & 1;
        const char* tiles = smem + cur * 3 * TILE1_BYTES;
        if (wid == 0) __builtin_amdgcn_s_wait_tensorcnt(0);
        __syncthreads();                      // tile 'cur' visible to all waves
        if (wid == 0 && (k0 + 32) < DIN) {    // prefetch next k-slice
            unsigned nb = lds_base + (cur ^ 1) * 3 * TILE1_BYTES;
            long long off = (long long)col0 * DIN + (k0 + 32);
            tdm_load_tile(nb + 0 * TILE1_BYTES, wqT, off, 32);
            tdm_load_tile(nb + 1 * TILE1_BYTES, wkT, off, 32);
            tdm_load_tile(nb + 2 * TILE1_BYTES, wvT, off, 32);
        }
        v16bf a = load_a_frag(xb, arow, k0 + lh * 8);
#pragma unroll
        for (int nt = 0; nt < 2; ++nt) {
            int nl = (nt << 4) + lm;          // local column within tile
            cq[nt] = wmma_bf16(a, lds_b_frag(tiles + 0 * TILE1_BYTES, nl, lh), cq[nt]);
            ck[nt] = wmma_bf16(a, lds_b_frag(tiles + 1 * TILE1_BYTES, nl, lh), ck[nt]);
            cv[nt] = wmma_bf16(a, lds_b_frag(tiles + 2 * TILE1_BYTES, nl, lh), cv[nt]);
        }
    }

    const int batch = m0 >> 13;               // 128 | 8192: block never crosses batch
#pragma unroll
    for (int nt = 0; nt < 2; ++nt) {
        int col = col0 + (nt << 4) + lm;
        float bqv = bq[col], bkv = bk[col], bvv = bv[col];
        int rbase = m0 + lh * 8;
        float skv = 0.0f, sk = 0.0f;
#pragma unroll
        for (int r = 0; r < 8; ++r) {
            float q  = cq[nt][r] + bqv; q  = (q  > 0.0f) ? (q  + 1.0f) : __expf(q);
            float kk = ck[nt][r] + bkv; kk = (kk > 0.0f) ? (kk + 1.0f) : __expf(kk);
            float vv = cv[nt][r] + bvv;
            qfeat[(size_t)(rbase + r) * H + col] = (__bf16)q;
            skv += kk * vv;
            sk  += kk;
        }
        skv += __shfl_xor(skv, 16);           // merge row halves (same column)
        sk  += __shfl_xor(sk, 16);
        if (lane < 16) {
            atomicAdd(&kvred[batch * H + col], skv);
            atomicAdd(&kred [batch * H + col], sk);
        }
    }
}

// ---------------- kernel 2: V' = Q*KV/(Q*Ksum+eps); out = gelu(V'@wo + bo) --
// Block = 8 waves = 128 rows x 64 cols; wo tiles staged by TDM double buffer.

__global__ __launch_bounds__(256) void out_wmma(
    const __bf16* __restrict__ qfeat,
    const float* __restrict__ kvred, const float* __restrict__ kred,
    const __bf16* __restrict__ woT, const float* __restrict__ bo,
    float* __restrict__ out)
{
    __shared__ alignas(16) char smem[2 * TILE2_BYTES];
    const int lane = threadIdx.x & 31;
    const int wid  = threadIdx.x >> 5;
    const int rg = blockIdx.x >> 4;          // 256 row groups of 128
    const int cg = blockIdx.x & 15;          // 16 col groups of 64
    const int m0 = (rg << 7) + (wid << 4);
    const int lm = lane & 15, lh = lane >> 4;
    const int arow = m0 + lm;
    const int col0 = cg << 6;
    const int batch = m0 >> 13;
    const float* kvb = kvred + batch * H;
    const float* ksb = kred  + batch * H;
    const unsigned lds_base = (unsigned)(size_t)&smem[0];

    v8f acc[4];
#pragma unroll
    for (int t = 0; t < 4; ++t) acc[t] = (v8f){0,0,0,0,0,0,0,0};

    if (wid == 0)
        tdm_load_tile(lds_base, woT, (long long)col0 * DIN, 64);

    for (int k0 = 0; k0 < H; k0 += 32) {
        const int cur = (k0 >> 5) & 1;
        const char* tile = smem + cur * TILE2_BYTES;
        if (wid == 0) __builtin_amdgcn_s_wait_tensorcnt(0);
        __syncthreads();
        if (wid == 0 && (k0 + 32) < H)
            tdm_load_tile(lds_base + (cur ^ 1) * TILE2_BYTES, woT,
                          (long long)col0 * DIN + (k0 + 32), 64);

        // Build A fragment: V' features for row arow, on the fly in f32
        int klo = k0 + lh * 8;
        FragU af;
#pragma unroll
        for (int run = 0; run < 2; ++run) {
            int kk = klo + run * 16;
            Pack8 qa;
            qa.u = *reinterpret_cast<const uint4*>(qfeat + (size_t)arow * H + kk);
            const float4* kvp = reinterpret_cast<const float4*>(kvb + kk);
            const float4* ksp = reinterpret_cast<const float4*>(ksb + kk);
            float4 kv0 = kvp[0], kv1 = kvp[1];
            float4 ks0 = ksp[0], ks1 = ksp[1];
            float kvv[8] = {kv0.x, kv0.y, kv0.z, kv0.w, kv1.x, kv1.y, kv1.z, kv1.w};
            float ksv[8] = {ks0.x, ks0.y, ks0.z, ks0.w, ks1.x, ks1.y, ks1.z, ks1.w};
            Pack8 va;
#pragma unroll
            for (int i = 0; i < 8; ++i) {
                float q = (float)qa.h[i];
                float r = __builtin_amdgcn_rcpf(q * ksv[i] + 1e-6f);
                va.h[i] = (__bf16)(q * kvv[i] * r);
            }
            af.u[run] = va.u;
        }
#pragma unroll
        for (int nt = 0; nt < 4; ++nt) {
            int nl = (nt << 4) + lm;
            acc[nt] = wmma_bf16(af.v, lds_b_frag(tile, nl, lh), acc[nt]);
        }
    }

#pragma unroll
    for (int nt = 0; nt < 4; ++nt) {
        int col = col0 + (nt << 4) + lm;
        float bov = bo[col];
        int rbase = m0 + lh * 8;
#pragma unroll
        for (int r = 0; r < 8; ++r) {
            float v = acc[nt][r] + bov;
            float u = 0.7978845608028654f * (v + 0.044715f * v * v * v);
            out[(size_t)(rbase + r) * H + col] = 0.5f * v * (1.0f + tanhf(u));
        }
    }
}

// ---------------- host launcher ---------------------------------------------

extern "C" void kernel_launch(void* const* d_in, const int* in_sizes, int n_in,
                              void* d_out, int out_size, void* d_ws, size_t ws_size,
                              hipStream_t stream) {
    const float* x  = (const float*)d_in[0];
    const float* wq = (const float*)d_in[1];
    const float* bq = (const float*)d_in[2];
    const float* wk = (const float*)d_in[3];
    const float* bk = (const float*)d_in[4];
    const float* wv = (const float*)d_in[5];
    const float* bv = (const float*)d_in[6];
    const float* wo = (const float*)d_in[7];
    const float* bo = (const float*)d_in[8];
    float* out = (float*)d_out;

    char* ws = (char*)d_ws;
    size_t off = 0;
    __bf16* xb    = (__bf16*)(ws + off); off += (size_t)NB * S_LEN * DIN * 2; // 64 MiB
    __bf16* wqT   = (__bf16*)(ws + off); off += (size_t)DIN * H * 2;
    __bf16* wkT   = (__bf16*)(ws + off); off += (size_t)DIN * H * 2;
    __bf16* wvT   = (__bf16*)(ws + off); off += (size_t)DIN * H * 2;
    __bf16* woT   = (__bf16*)(ws + off); off += (size_t)DIN * H * 2;
    __bf16* qfeat = (__bf16*)(ws + off); off += (size_t)NB * S_LEN * H * 2;   // 64 MiB
    float*  kvred = (float*) (ws + off); off += (size_t)NB * H * 4;
    float*  kred  = (float*) (ws + off); off += (size_t)NB * H * 4;

    zero_f32<<<(2 * NB * H + 255) / 256, 256, 0, stream>>>(kvred, 2 * NB * H);

    size_t n8 = (size_t)NB * S_LEN * DIN / 8;
    cvt_x_bf16<<<(unsigned)((n8 + 255) / 256), 256, 0, stream>>>(x, xb, n8);

    dim3 tb(32, 8), tg(H / 32, DIN / 32);
    transpose_cvt<<<tg, tb, 0, stream>>>(wq, wqT);
    transpose_cvt<<<tg, tb, 0, stream>>>(wk, wkT);
    transpose_cvt<<<tg, tb, 0, stream>>>(wv, wvT);
    transpose_cvt<<<tg, tb, 0, stream>>>(wo, woT);

    // kernel 1: 256 row groups x 32 col groups
    qkv_wmma<<<8192, 256, 0, stream>>>(xb, wqT, wkT, wvT, bq, bk, bv,
                                       qfeat, kvred, kred);

    // kernel 2: 256 row groups x 16 col groups
    out_wmma<<<4096, 256, 0, stream>>>(qfeat, kvred, kred, woT, bo, out);
}
